// All_Hausdorff_Distances_Chunked_35416300323042
// MI455X (gfx1250) — compile-verified, matrix operand order checked by reading, not run on hardware
//
#include <hip/hip_runtime.h>
#include <hip/hip_bf16.h>

typedef __attribute__((ext_vector_type(16))) _Float16 v16h;
typedef __attribute__((ext_vector_type(8)))  float    v8f;
typedef __attribute__((ext_vector_type(2)))  _Float16 h2;
typedef __attribute__((ext_vector_type(8)))  unsigned u32x8;

#define HD_B 4
#define HD_C 4
#define HD_H 96
#define HD_W 96
#define HD_N (HD_H * HD_W)          // 9216
#define HD_NPAIR (HD_B * HD_C)      // 16
#define HD_BIG 1e30f
#define HD_D2MAX 18050              // 2*95*95
#define HD_CTILES (HD_N / 16)       // 576
#define HD_RTILES (HD_N / 16)       // 576
#define HD_TILES_PER_BLOCK 8        // 8 waves of 32 = 256 threads
#define HD_BLOCKS_PER_TASK (HD_RTILES / HD_TILES_PER_BLOCK) // 72

// ---- workspace layout (floats) ----
// [0,        16N)  offA[p][i]  : 0 if pred-class mask, else 1e30
// [16N,      32N)  offB[p][i]  : 0 if label mask, else 1e30
// [32N,      48N)  edtB[p][i]  : d2min to maskB set (float; 1e30 if empty)  -> forward
// [48N,      64N)  edtA[p][i]  : d2min to maskA set                         -> reverse
// [64N, 64N+96)    fr[p][dir][stat] : post-fallback forward/reverse stats
#define WS_OFFA 0
#define WS_OFFB (16 * HD_N)
#define WS_EDTB (32 * HD_N)
#define WS_EDTA (48 * HD_N)
#define WS_FR   (64 * HD_N)

// ------------------------------------------------------------------
// Kernel 1: build mask-offset arrays (0 = in set, 1e30 = not in set)
// ------------------------------------------------------------------
__global__ __launch_bounds__(256) void hd_prep(const float* __restrict__ pred,
                                               const float* __restrict__ lab,
                                               float* __restrict__ ws) {
  int t = blockIdx.x * 256 + threadIdx.x;
  if (t >= HD_B * HD_N) return;
  int b = t / HD_N;
  int i = t - b * HD_N;
  const float* pb = pred + (size_t)b * HD_C * HD_N + i;
  float best = pb[0];
  int am = 0;
  #pragma unroll
  for (int c = 1; c < HD_C; ++c) {
    float v = pb[(size_t)c * HD_N];
    if (v > best) { best = v; am = c; }   // first-occurrence argmax (strict >)
  }
  const float* lb = lab + (size_t)b * HD_C * HD_N + i;
  #pragma unroll
  for (int c = 0; c < HD_C; ++c) {
    ws[WS_OFFA + (b * HD_C + c) * HD_N + i] = (c == am) ? 0.0f : HD_BIG;
    ws[WS_OFFB + (b * HD_C + c) * HD_N + i] = (lb[(size_t)c * HD_N] > 0.5f) ? 0.0f : HD_BIG;
  }
}

// ------------------------------------------------------------------
// Kernel 2: WMMA masked-min "EDT": for each of 32 tasks (16 pairs x
// {maskB, maskA}), d2min[i] = min_{j in mask} d2(i,j), via
// d2 = sq_i + (sq_j + off_j) - 2*(y_i*y_j + x_i*x_j)
// K=2 inner products computed 16x16 per v_wmma_f32_16x16x32_f16 tile.
// LDS stages, per column j, a uint2 {tj = sq_j + off_j (f32), packed
// half2(y_j, x_j)} so the inner loop is one ds_load_b64 + one v_and
// + WMMA + fma/min per tile. sq_i is hoisted out of the min.
// ------------------------------------------------------------------
__global__ __launch_bounds__(256) void hd_edt(float* __restrict__ ws) {
  __shared__ uint2 sPack[HD_N];   // 72 KB: {tj, half2(y,x)} per column

  int task  = blockIdx.x / HD_BLOCKS_PER_TASK;   // 0..31
  int chunk = blockIdx.x - task * HD_BLOCKS_PER_TASK;
  int p   = task >> 1;
  int src = task & 1;           // 0: maskB -> edtB (forward), 1: maskA -> edtA (reverse)
  const float* msk = ws + (src == 0 ? WS_OFFB : WS_OFFA) + p * HD_N;
  float*       out = ws + (src == 0 ? WS_EDTB : WS_EDTA) + p * HD_N;

  int tid = threadIdx.x;
  for (int idx = tid; idx < HD_N; idx += 256) {
    __builtin_prefetch(msk + idx + 1024, 0, 0);   // global_prefetch_b8 (speculative)
    int y = idx / HD_W;
    int x = idx - y * HD_W;
    float tj = (float)(y * y + x * x) + msk[idx];
    h2 yx;
    yx[0] = (_Float16)y;
    yx[1] = (_Float16)x;
    uint2 pk;
    pk.x = __builtin_bit_cast(unsigned, tj);
    pk.y = __builtin_bit_cast(unsigned, yx);
    sPack[idx] = pk;
  }
  __syncthreads();

  int lane = tid & 31;
  int wv   = tid >> 5;
  int rowbase = (chunk * HD_TILES_PER_BLOCK + wv) * 16;
  bool lo16 = (lane < 16);
  int half = lane >> 4;
  unsigned lanemask = lo16 ? 0xFFFFFFFFu : 0u;   // zero K=0,1 for lanes 16-31

  // A fragment: 16x32 f16, rows M = lane (lanes 0-15), K0=y_i, K1=x_i, rest 0.
  u32x8 aw = {};
  {
    int i = rowbase + (lane & 15);
    int yi = i / HD_W;
    int xi = i - yi * HD_W;
    h2 ah;
    ah[0] = (_Float16)yi;
    ah[1] = (_Float16)xi;
    aw[0] = __builtin_bit_cast(unsigned, ah) & lanemask;
  }
  v16h afrag = __builtin_bit_cast(v16h, aw);

  // sq_i for output rows this lane owns (D layout: VGPR v -> M = v + 8*half).
  float sqi[8];
  #pragma unroll
  for (int v = 0; v < 8; ++v) {
    int i = rowbase + v + half * 8;
    int y = i / HD_W;
    int x = i - y * HD_W;
    sqi[v] = (float)(y * y + x * x);
  }

  v8f acc;
  #pragma unroll
  for (int v = 0; v < 8; ++v) acc[v] = 3.0e38f;
  v8f czero = {};

  u32x8 bw = {};                       // B operand words; only word 0 varies
  const uint2* pj = sPack + (lane & 15);

  for (int it = 0; it < HD_CTILES; ++it) {
    uint2 pk = *pj;
    pj += 16;
    float tj = __builtin_bit_cast(float, pk.x);  // sq_j + off_j (1e30 absorbs sq_j)
    bw[0] = pk.y & lanemask;                     // half2(y_j, x_j), zeroed for hi lanes
    v16h bfrag = __builtin_bit_cast(v16h, bw);

    v8f dot = __builtin_amdgcn_wmma_f32_16x16x32_f16(
        /*neg_a=*/false, afrag, /*neg_b=*/false, bfrag,
        /*c_mod=*/(short)0, czero, /*reuse_a=*/false, /*reuse_b=*/false);

    #pragma unroll
    for (int v = 0; v < 8; ++v) {
      float u = fmaf(dot[v], -2.0f, tj);  // (sq_j + off_j) - 2*dot (exact ints / exact 1e30)
      acc[v] = fminf(acc[v], u);
    }
  }

  // min across the 16 lanes sharing each output row, then add sq_i.
  #pragma unroll
  for (int v = 0; v < 8; ++v) {
    float m = acc[v];
    m = fminf(m, __shfl_xor(m, 1, 32));
    m = fminf(m, __shfl_xor(m, 2, 32));
    m = fminf(m, __shfl_xor(m, 4, 32));
    m = fminf(m, __shfl_xor(m, 8, 32));
    acc[v] = m + sqi[v];                    // empty mask -> exactly 1e30
  }
  if ((lane & 15) == 0) {
    #pragma unroll
    for (int v = 0; v < 8; ++v) out[rowbase + v + half * 8] = acc[v];
  }
}

// ------------------------------------------------------------------
// Kernel 3: per-pair stats. One block per pair; both directions.
// Order statistics via uniform binary search over integer d2 (<=18050).
// ------------------------------------------------------------------
__global__ __launch_bounds__(256) void hd_stats(float* __restrict__ ws) {
  __shared__ int   keys[HD_N];   // 36 KB: d2 for valid&finite, INT_MAX otherwise
  __shared__ int   sredi[256];
  __shared__ float sredf[256];

  int p = blockIdx.x;
  int tid = threadIdx.x;
  float outst[2][3];
  int nA_saved = 0;

  for (int dir = 0; dir < 2; ++dir) {
    const float* val = ws + (dir == 0 ? WS_EDTB : WS_EDTA) + p * HD_N;
    const float* msk = ws + (dir == 0 ? WS_OFFA : WS_OFFB) + p * HD_N;

    int cnt = 0, cntf = 0;
    float sum = 0.0f, mx = -HD_BIG;
    for (int i = tid; i < HD_N; i += 256) {
      bool valid = (msk[i] == 0.0f);
      float a = val[i];
      bool big = (a >= 1e29f);
      float d = big ? HD_BIG : sqrtf(a);
      int key = 0x7FFFFFFF;
      if (valid) {
        cnt++;
        sum += d;
        mx = fmaxf(mx, d);
        if (!big) { cntf++; key = (int)a; }
      }
      keys[i] = key;
    }
    __syncthreads();

    // reductions: n, nfinite, sum, max
    sredi[tid] = cnt; __syncthreads();
    for (int s = 128; s > 0; s >>= 1) { if (tid < s) sredi[tid] += sredi[tid + s]; __syncthreads(); }
    int n = sredi[0]; __syncthreads();

    sredi[tid] = cntf; __syncthreads();
    for (int s = 128; s > 0; s >>= 1) { if (tid < s) sredi[tid] += sredi[tid + s]; __syncthreads(); }
    int nfin = sredi[0]; __syncthreads();

    sredf[tid] = sum; __syncthreads();
    for (int s = 128; s > 0; s >>= 1) { if (tid < s) sredf[tid] += sredf[tid + s]; __syncthreads(); }
    float S = sredf[0]; __syncthreads();

    sredf[tid] = mx; __syncthreads();
    for (int s = 128; s > 0; s >>= 1) { if (tid < s) sredf[tid] = fmaxf(sredf[tid], sredf[tid + s]); __syncthreads(); }
    float MX = sredf[0]; __syncthreads();

    auto count_le = [&](int T) -> int {
      int c = 0;
      for (int i = tid; i < HD_N; i += 256) c += (keys[i] <= T) ? 1 : 0;
      sredi[tid] = c; __syncthreads();
      for (int s = 128; s > 0; s >>= 1) { if (tid < s) sredi[tid] += sredi[tid + s]; __syncthreads(); }
      int r = sredi[0]; __syncthreads();
      return r;
    };
    auto kth = [&](int k) -> float {        // k-th (0-indexed) order stat of where(mask, dmin, BIG)
      if (k + 1 > nfin) return HD_BIG;      // falls into the 1e30 region
      int loT = 0, hiT = HD_D2MAX;
      while (loT < hiT) {                   // uniform control flow: bounds are block-uniform
        int mid = (loT + hiT) >> 1;
        int c = count_le(mid);
        if (c >= k + 1) hiT = mid; else loT = mid + 1;
      }
      return sqrtf((float)loT);
    };

    float mean = S / (float)((n > 1) ? n : 1);
    float p95;
    if (n == 0) {
      p95 = HD_BIG;                         // s[0] of all-BIG array
    } else {
      float nf = (float)(n - 1);
      float idx = 0.95f * nf;
      int lo = (int)floorf(idx);
      int hi = (int)ceilf(idx);
      lo = lo < 0 ? 0 : (lo > HD_N - 1 ? HD_N - 1 : lo);
      hi = hi < 0 ? 0 : (hi > HD_N - 1 ? HD_N - 1 : hi);
      float frac = idx - (float)lo;
      float slo = kth(lo);
      float shi = (hi == lo) ? slo : kth(hi);
      p95 = slo * (1.0f - frac) + shi * frac;
    }

    outst[dir][0] = MX;
    outst[dir][1] = mean;
    outst[dir][2] = p95;
    if (dir == 0) nA_saved = n;
    __syncthreads();                        // before keys[] is reused
  }

  if (tid == 0) {
    float fill = (float)((HD_H + HD_W) / 4);   // 48
    float* fr = ws + WS_FR + p * 6;
    #pragma unroll
    for (int s = 0; s < 3; ++s) {
      float f = outst[0][s], r = outst[1][s];
      if (nA_saved == 0) { f = fill; r = fill; }
      fr[s] = f;
      fr[3 + s] = r;
    }
  }
}

// ------------------------------------------------------------------
// Kernel 4: finish -> (MHD, FHD, RHD) each [B, STATS, C+2] = [4,3,6]
// ------------------------------------------------------------------
__global__ __launch_bounds__(256) void hd_finish(const float* __restrict__ ws,
                                                 float* __restrict__ outp) {
  int t = threadIdx.x;
  if (t >= 3 * HD_B * 3 * 6) return;        // 216
  const float* fr = ws + WS_FR;
  int o = t / 72;
  int rem = t - o * 72;
  int b = rem / 18;
  int rem2 = rem - b * 18;
  int s = rem2 / 6;
  int col = rem2 - s * 6;

  float x[HD_C];
  #pragma unroll
  for (int c = 0; c < HD_C; ++c) {
    int p = b * HD_C + c;
    float f = fr[p * 6 + s];
    float r = fr[p * 6 + 3 + s];
    float v = (o == 0) ? fmaxf(f, r) : ((o == 1) ? f : r);
    x[c] = (c == 0) ? 0.0f : v;             // IGNORE_CLASS = 0
  }
  float res;
  if (col < 4)       res = x[col];
  else if (col == 4) res = (x[0] + x[1] + x[2] + x[3]) * 0.25f;
  else               res = (x[1] + x[2] + x[3]) * (1.0f / 3.0f);
  outp[t] = res;
}

// ------------------------------------------------------------------
extern "C" void kernel_launch(void* const* d_in, const int* in_sizes, int n_in,
                              void* d_out, int out_size, void* d_ws, size_t ws_size,
                              hipStream_t stream) {
  const float* pred = (const float*)d_in[0];
  const float* lab  = (const float*)d_in[1];
  float* ws  = (float*)d_ws;
  float* out = (float*)d_out;

  hd_prep<<<(HD_B * HD_N + 255) / 256, 256, 0, stream>>>(pred, lab, ws);
  hd_edt<<<32 * HD_BLOCKS_PER_TASK, 256, 0, stream>>>(ws);
  hd_stats<<<HD_NPAIR, 256, 0, stream>>>(ws);
  hd_finish<<<1, 256, 0, stream>>>(ws, out);
}
